// PostNormBoth_51823075394177
// MI455X (gfx1250) — matrix operand order repaired and checked by reading, not compiled, not run on hardware
//
#include <hip/hip_runtime.h>
#include <hip/hip_bf16.h>
#include <math.h>

#define HID   256
#define SLOTS 64
#define TT    256
#define ROWS  16
#define TAUF  8.0f
#define EPSF  1e-5f

typedef __attribute__((ext_vector_type(16))) __bf16 v16bf;
typedef __attribute__((ext_vector_type(8)))  float  v8f;

union FragAB { v16bf v; unsigned int u[8]; uint4 q[2]; };
union Pack4  { __bf16 h[4]; uint2 u2; };

// Dynamic LDS layout (bytes)
#define OFF_W    0                         // W_update bf16 [256][256]      128 KB
#define OFF_A    131072                    // A tile  bf16  [16][256]         8 KB
#define OFF_H    139264                    // pre-norm h f32 [16][256]       16 KB
#define OFF_HC   155648                    // carry   h f32 [16][256]        16 KB
#define OFF_ST   172032                    // layernorm stats f32 [16][2]
#define SMEM_BYTES 172160

__global__ __launch_bounds__(256)
void recurrent_wmma_kernel(const float* __restrict__ x,
                           const float* __restrict__ W_embed,
                           const float* __restrict__ b_embed,
                           const float* __restrict__ W_update,
                           const float* __restrict__ b_update,
                           const float* __restrict__ gamma,
                           const float* __restrict__ beta,
                           const float* __restrict__ W_out,
                           const float* __restrict__ b_out,
                           const float* __restrict__ cs_in,
                           float* __restrict__ out,
                           float* __restrict__ mem_ws)
{
    extern __shared__ char smem[];
    __bf16* Wlds = (__bf16*)(smem + OFF_W);
    __bf16* Alds = (__bf16*)(smem + OFF_A);
    float*  Hlds = (float*)(smem + OFF_H);
    float*  hlds = (float*)(smem + OFF_HC);
    float*  stat = (float*)(smem + OFF_ST);

    const int tid  = threadIdx.x;          // 0..255 (== HID)
    const int lane = tid & 31;
    const int wid  = tid >> 5;             // 0..7
    const int lo16 = lane & 15;
    const int hi   = lane >> 4;            // 0/1
    const int b0   = blockIdx.x * ROWS;

    float* memblk = mem_ws + (size_t)blockIdx.x * ROWS * SLOTS * HID;

    // ---- init per call (deterministic): W -> bf16 LDS, zero h and memory ----
    for (int i = tid * 4; i < HID * HID; i += 1024) {
        float4 wv = *(const float4*)&W_update[i];
        Pack4 p;
        p.h[0] = (__bf16)wv.x; p.h[1] = (__bf16)wv.y;
        p.h[2] = (__bf16)wv.z; p.h[3] = (__bf16)wv.w;
        *(uint2*)&Wlds[i] = p.u2;                       // ds_store_b64
    }
    for (int i = tid; i < ROWS * HID; i += 256) hlds[i] = 0.0f;
    {
        const float4 z4 = make_float4(0.f, 0.f, 0.f, 0.f);
        for (int i = tid * 4; i < ROWS * SLOTS * HID; i += 1024)
            *(float4*)&memblk[i] = z4;                  // global_store_b128
    }

    const float we  = W_embed[tid];
    const float be  = b_embed[tid];
    const float g   = gamma[tid];
    const float bt  = beta[tid];
    const float cs  = 1.0f / (1.0f + __expf(-cs_in[0]));

    // Each wave owns N-tiles [32*wid, 32*wid+32); per-lane columns fixed for all steps.
    const int   n0  = wid * 32 + lo16;
    const int   n1  = n0 + 16;
    const float bu0 = b_update[n0];
    const float bu1 = b_update[n1];

    __syncthreads();

    for (int t = 0; t < TT; ++t) {
        // ---- attention indices/weights: pure function of t (pointer = t mod 64) ----
        const int base = t & (SLOTS - 1);
        int   idxs[5];
        float wts[5];
        float ssum = 0.f;
        #pragma unroll
        for (int k = 0; k < 5; ++k) {
            int ii   = (base + k - 2 + SLOTS) & (SLOTS - 1);
            idxs[k]  = ii;
            float d  = (float)ii - (float)base;
            float e  = __expf(-d * d * (1.0f / TAUF));
            wts[k] = e; ssum += e;
        }
        const float inv = 1.0f / ssum;
        #pragma unroll
        for (int k = 0; k < 5; ++k) wts[k] *= inv;

        // ---- phase 1: A = inp + cs*context + h_prev  -> bf16 LDS tile ----
        for (int m = 0; m < ROWS; ++m) {
            float xv  = x[(b0 + m) * TT + t];
            float inp = tanhf(xv * we + be);
            const float* mp = memblk + (m * SLOTS) * HID + tid;
            float ctx = 0.f;
            #pragma unroll
            for (int k = 0; k < 5; ++k) ctx += wts[k] * mp[idxs[k] * HID];
            float a = inp + cs * ctx + hlds[m * HID + tid];
            Alds[m * HID + tid] = (__bf16)a;
        }
        __syncthreads();

        // ---- phase 2: [16x256] @ W^T via v_wmma_f32_16x16x32_bf16 ----
        // Fragments are two contiguous 16B runs per lane -> ds_load_b128 pairs.
        v8f acc0 = {};
        v8f acc1 = {};
        #pragma unroll
        for (int kc = 0; kc < 8; ++kc) {
            const int k0 = kc * 32;
            FragAB af, b0f, b1f;
            // A 16x32 bf16: lanes 0-15 K=0..15, lanes 16-31 K=16..31 (contiguous 32B).
            const __bf16* ab = Alds + lo16 * HID + k0 + hi * 16;
            af.q[0] = *(const uint4*)(ab);          // k: +0..7
            af.q[1] = *(const uint4*)(ab + 8);      // k: +8..15
            // B 32x16 bf16: lanes 0-15 K=0..15, lanes 16-31 K=16..31 (contiguous 32B).
            const __bf16* bb0 = Wlds + n0 * HID + k0 + hi * 16;
            b0f.q[0] = *(const uint4*)(bb0);
            b0f.q[1] = *(const uint4*)(bb0 + 8);
            const __bf16* bb1 = Wlds + n1 * HID + k0 + hi * 16;
            b1f.q[0] = *(const uint4*)(bb1);
            b1f.q[1] = *(const uint4*)(bb1 + 8);

            acc0 = __builtin_amdgcn_wmma_f32_16x16x32_bf16(false, af.v, false, b0f.v,
                                                           (short)0, acc0, false, false);
            acc1 = __builtin_amdgcn_wmma_f32_16x16x32_bf16(false, af.v, false, b1f.v,
                                                           (short)0, acc1, false, false);
        }

        // ---- phase 3: bias + tanh -> pre-norm h in LDS ----
        #pragma unroll
        for (int r = 0; r < 8; ++r) {
            int m = r + hi * 8;                 // C/D layout: VGPR r -> row r (+8 for hi half)
            Hlds[m * HID + n0] = tanhf(acc0[r] + bu0);
            Hlds[m * HID + n1] = tanhf(acc1[r] + bu1);
        }
        __syncthreads();

        // ---- phase 4a: layernorm stats (16 lanes per row, wave32 shfl width 16) ----
        {
            int mrow = tid >> 4;
            int s    = tid & 15;
            float sum = 0.f, sq = 0.f;
            #pragma unroll
            for (int j = 0; j < 16; ++j) {
                float v = Hlds[mrow * HID + s + 16 * j];
                sum += v; sq += v * v;
            }
            #pragma unroll
            for (int msk = 8; msk >= 1; msk >>= 1) {
                sum += __shfl_xor(sum, msk, 16);
                sq  += __shfl_xor(sq,  msk, 16);
            }
            float mu  = sum * (1.0f / HID);
            float var = sq * (1.0f / HID) - mu * mu;
            float rs  = rsqrtf(var + EPSF);
            if (s == 0) { stat[mrow * 2] = mu; stat[mrow * 2 + 1] = rs; }
        }
        __syncthreads();

        // ---- phase 4b: normalize, write carry h, exclusive-owner scatter-add ----
        for (int m = 0; m < ROWS; ++m) {
            float mu = stat[m * 2], rs = stat[m * 2 + 1];
            float hn = (Hlds[m * HID + tid] - mu) * rs * g + bt;
            hlds[m * HID + tid] = hn;
            float* mp = memblk + (m * SLOTS) * HID + tid;
            #pragma unroll
            for (int k = 0; k < 5; ++k)
                mp[idxs[k] * HID] += wts[k] * hn;   // (m, tid) owned by this thread only
        }
        __syncthreads();
    }

    // ---- epilogue: out[b,:10] = h @ W_out^T + b_out ----
    if (tid < ROWS * 10) {
        int m = tid / 10, o = tid % 10;
        const float* wrow = W_out + o * HID;
        const float* hr   = hlds + m * HID;
        float s = b_out[o];
        for (int c = 0; c < HID; ++c) s += hr[c] * wrow[c];
        out[(b0 + m) * 10 + o] = s;
    }
}

extern "C" void kernel_launch(void* const* d_in, const int* in_sizes, int n_in,
                              void* d_out, int out_size, void* d_ws, size_t ws_size,
                              hipStream_t stream) {
    const float* x       = (const float*)d_in[0];
    const float* W_embed = (const float*)d_in[1];
    const float* b_embed = (const float*)d_in[2];
    const float* W_upd   = (const float*)d_in[3];
    const float* b_upd   = (const float*)d_in[4];
    const float* gamma   = (const float*)d_in[5];
    const float* beta    = (const float*)d_in[6];
    const float* W_out   = (const float*)d_in[7];
    const float* b_out   = (const float*)d_in[8];
    const float* cstr    = (const float*)d_in[9];
    float* out           = (float*)d_out;

    const int B    = in_sizes[0] / TT;     // 512
    const int grid = B / ROWS;             // 32 workgroups, 16 batch rows each

    // Allow >64KB dynamic LDS (gfx1250 WGP has 320KB); ignore failure.
    (void)hipFuncSetAttribute(reinterpret_cast<const void*>(recurrent_wmma_kernel),
                              hipFuncAttributeMaxDynamicSharedMemorySize, SMEM_BYTES);

    recurrent_wmma_kernel<<<grid, 256, SMEM_BYTES, stream>>>(
        x, W_embed, b_embed, W_upd, b_upd, gamma, beta, W_out, b_out, cstr,
        out, (float*)d_ws);
}
